// RecurrentApproximator_27633819583179
// MI455X (gfx1250) — matrix operand order repaired and checked by reading, not compiled, not run on hardware
//
#include <hip/hip_runtime.h>
#include <hip/hip_bf16.h>

// ---------------------------------------------------------------------------
// Seq2seq RNN + attention for MI455X (gfx1250), bf16 WMMA core.
//   - GEMMs via v_wmma_f32_16x16x32_bf16 (f32 accumulate, f32 bias+tanh).
//   - Fragment-swizzled bf16 weight image, streamed from L2 each step.
//     A laundered zero *offset* (not pointer) blocks loop-invariant hoisting
//     while preserving global address-space provenance -> global_load_b128.
//   - Activations staged in LDS (buffer select by arithmetic, so fragment
//     reads stay ds_load_b128); TDM (tensor_load_to_lds) double-buffers the
//     next time-step / next layer tile while WMMAs execute; s_wait_tensorcnt.
//   - Sequential phases: one 1024-thread WG (32 waves);
//     parallel phases (embed, u_keys, scores, softmax+ctx): full grid.
// ---------------------------------------------------------------------------

typedef __attribute__((ext_vector_type(16))) __bf16 v16bf;
typedef __attribute__((ext_vector_type(8)))  float  v8f;

#define Hd   384
#define Bsz  64
#define Tsrc 256
#define Tdec 255
#define Lnum 6

#define TDM_OK __has_builtin(__builtin_amdgcn_tensor_load_to_lds)

// ------------------------- offset laundering -------------------------------
// Opaque zero: defeats loop-invariant hoisting of weight-fragment loads
// (which previously preloaded all fragments and spilled to scratch) while
// keeping the base pointer's AS(1) provenance intact.
__device__ __forceinline__ size_t launder_zero() {
  size_t off = 0;
  asm volatile("" : "+s"(off));
  return off;
}

// ------------------------------ TDM helpers --------------------------------
#if TDM_OK
typedef unsigned int u32x4 __attribute__((ext_vector_type(4)));
typedef int          i32x4 __attribute__((ext_vector_type(4)));
typedef int          i32x8 __attribute__((ext_vector_type(8)));

// 2D tile load: rows x rowUnits (8-byte units), global row stride strideUnits.
__device__ __forceinline__ void tdm_load_2d(unsigned int ldsOff, const void* g,
                                            int rowUnits, int rows,
                                            int strideUnits) {
  const unsigned long long ga = (unsigned long long)(uintptr_t)g;
  u32x4 g0;
  g0[0] = 1u;                                           // count=1, flags=0
  g0[1] = ldsOff;                                       // lds_addr
  g0[2] = (unsigned int)ga;                             // global_addr lo
  g0[3] = ((unsigned int)(ga >> 32) & 0x1FFFFFFu) | (2u << 30);  // hi | type=2
  i32x8 g1;
  g1[0] = (3 << 16);                                    // mask=0, data_size=8B
  g1[1] = (rowUnits & 0xFFFF) << 16;                    // tensor_dim0 lo
  g1[2] = ((rowUnits >> 16) & 0xFFFF) | ((rows & 0xFFFF) << 16);  // d0hi|d1lo
  g1[3] = ((rows >> 16) & 0xFFFF) | ((rowUnits & 0xFFFF) << 16);  // d1hi|tile0
  g1[4] = rows & 0xFFFF;                                // tile_dim1, tile_dim2=0
  g1[5] = strideUnits;                                  // tensor_dim0_stride lo
  g1[6] = 0;
  g1[7] = 0;
  i32x4 z4 = {0, 0, 0, 0};
#if defined(__clang_major__) && __clang_major__ >= 23
  i32x8 z8 = {0, 0, 0, 0, 0, 0, 0, 0};
  __builtin_amdgcn_tensor_load_to_lds(g0, g1, z4, z4, z8, 0);
#else
  __builtin_amdgcn_tensor_load_to_lds(g0, g1, z4, z4, 0);
#endif
}
#endif  // TDM_OK

__device__ __forceinline__ unsigned int lds_offset(const void* p) {
  return (unsigned int)(uintptr_t)p;  // generic->LDS: low 32 bits = LDS offset
}

// Stage one time-step tile x_t (64 rows x 384 bf16, row stride Tsrc*Hd elems).
__device__ __forceinline__ void stage_x(__bf16* dst, const __bf16* src, int t) {
#if TDM_OK
  if (threadIdx.x < 32)
    tdm_load_2d(lds_offset(dst), (const void*)(src + (size_t)t * Hd),
                /*rowUnits=*/Hd * 2 / 8, /*rows=*/Bsz,
                /*strideUnits=*/Tsrc * Hd / 4);
#else
  for (int i = threadIdx.x; i < Bsz * 48; i += 1024) {
    const int b = i / 48, c = i % 48;
    ((uint4*)(dst + b * Hd))[c] =
        ((const uint4*)(src + ((size_t)b * Tsrc + t) * Hd))[c];
  }
#endif
}

// Stage one hidden slab h_l (64x384 bf16, contiguous 48KB).
__device__ __forceinline__ void stage_h(__bf16* dst, const __bf16* hidden,
                                        int l) {
#if TDM_OK
  if (threadIdx.x < 32)
    tdm_load_2d(lds_offset(dst), (const void*)(hidden + (size_t)l * Bsz * Hd),
                /*rowUnits=*/Bsz * Hd * 2 / 8, /*rows=*/1,
                /*strideUnits=*/Bsz * Hd / 4);
#else
  for (int i = threadIdx.x; i < Bsz * 48; i += 1024)
    ((uint4*)dst)[i] = ((const uint4*)(hidden + (size_t)l * Bsz * Hd))[i];
#endif
}

__device__ __forceinline__ void stage_wait() {
#if TDM_OK
  if (threadIdx.x < 32) {
#if __has_builtin(__builtin_amdgcn_s_wait_tensorcnt)
    __builtin_amdgcn_s_wait_tensorcnt(0);
#else
    asm volatile("s_wait_tensorcnt 0x0" ::: "memory");
#endif
  }
#endif
  __syncthreads();
}

// -------------------------- WMMA workgroup GEMM ----------------------------
// blockDim.x == 1024 (32 waves).  acc(64x384) += A(64x(KT*32)) * Wsw-slab.
// Swizzled block index = nt*KTfull + kt0 + kt  (K-slab split support).
__device__ __forceinline__ void wg_gemm_part(const __bf16* A, int lda,
                                             const __bf16* __restrict__ Wsw,
                                             int kt0, int KT, int KTfull,
                                             v8f acc[3]) {
  const int tid  = threadIdx.x;
  const int wave = tid >> 5, lane = tid & 31;
  const int rtile = wave >> 3;        // 0..3  (16-row tile)
  const int cgrp  = wave & 7;         // 0..7  (3 col tiles each)
  const int half  = lane >> 4, lo = lane & 15;
  const __bf16* Arow = A + (rtile * 16 + lo) * lda;
  for (int kt = 0; kt < KT; ++kt) {
    v16bf a;
    ((uint4*)&a)[0] = *(const uint4*)(Arow + kt * 32 + half * 8);
    ((uint4*)&a)[1] = *(const uint4*)(Arow + kt * 32 + 16 + half * 8);
#pragma unroll
    for (int j = 0; j < 3; ++j) {
      const int nt = cgrp * 3 + j;
      const __bf16* bp =
          Wsw + (((size_t)(nt * KTfull + kt0 + kt)) * 32 + lane) * 16;
      if (kt + 1 < KT)  // keep next K-slab of this column streaming from L2
        __builtin_prefetch(bp + 512, 0, 1);
      v16bf bfrag;
      ((uint4*)&bfrag)[0] = *(const uint4*)(bp);
      ((uint4*)&bfrag)[1] = *(const uint4*)(bp + 8);
      acc[j] = __builtin_amdgcn_wmma_f32_16x16x32_bf16(
          false, a, false, bfrag, (short)0, acc[j], false, false);
    }
  }
}

template <class F>
__device__ __forceinline__ void wg_epilogue(const v8f acc[3],
                                            const float* __restrict__ bias,
                                            F&& store) {
  const int tid  = threadIdx.x;
  const int wave = tid >> 5, lane = tid & 31;
  const int rtile = wave >> 3, cgrp = wave & 7;
  const int half  = lane >> 4, lo = lane & 15;
#pragma unroll
  for (int j = 0; j < 3; ++j) {
    const int n  = (cgrp * 3 + j) * 16 + lo;
    const float bv = bias ? bias[n] : 0.0f;
#pragma unroll
    for (int r = 0; r < 8; ++r) {
      const int m = rtile * 16 + half * 8 + r;
      store(m, n, acc[j][r] + bv);
    }
  }
}

// ------------------------------ prep kernels -------------------------------
// Build swizzled bf16 weight image for W = [W1^T ; W2^T] (K = K1+K2 rows).
__global__ void swizzle_kernel(const float* __restrict__ W1,
                               const float* __restrict__ W2,
                               int K1, int K2, int N, __bf16* __restrict__ dst) {
  const int K = K1 + K2, KT = K >> 5;
  const int total = K * N;
  for (int idx = blockIdx.x * blockDim.x + threadIdx.x; idx < total;
       idx += gridDim.x * blockDim.x) {
    const int j = idx & 15, L = (idx >> 4) & 31, blk = idx >> 9;
    const int kt = blk % KT, nt = blk / KT;
    const int n = nt * 16 + (L & 15);
    const int k = kt * 32 + (j & 8) * 2 + (L >> 4) * 8 + (j & 7);
    const float v = (k < K1) ? W1[(size_t)n * K1 + k]
                             : W2[(size_t)n * K2 + (k - K1)];
    dst[idx] = (__bf16)v;
  }
}

__global__ void bias_add_kernel(const float* __restrict__ a,
                                const float* __restrict__ b,
                                float* __restrict__ o, int n) {
  const int i = blockIdx.x * blockDim.x + threadIdx.x;
  if (i < n) o[i] = a[i] + b[i];
}

__global__ void embed_kernel(const float* __restrict__ inp,
                             const float* __restrict__ eW,
                             const float* __restrict__ eB,
                             __bf16* __restrict__ x0) {
  const size_t idx = (size_t)blockIdx.x * blockDim.x + threadIdx.x;
  if (idx >= (size_t)Bsz * Tsrc * Hd) return;
  const int h = (int)(idx % Hd);
  const size_t bt = idx / Hd;
  x0[idx] = (__bf16)(inp[bt] * eW[h] + eB[h]);
}

// --------------------------- encoder layer scan ----------------------------
// One WG, loops t:  h = tanh(x_t@Wih^T + h@Whh^T + bias), TDM double-buffer x.
__global__ void __launch_bounds__(1024)
enc_layer_kernel(const __bf16* __restrict__ xin, __bf16* __restrict__ yout,
                 const __bf16* __restrict__ Wsw, const float* __restrict__ bias,
                 __bf16* __restrict__ hfin) {
  extern __shared__ char smem[];
  __bf16* Xb0 = (__bf16*)smem;            // 64x384 (double buffered x2)
  __bf16* Hb  = Xb0 + 2 * Bsz * Hd;       // 64x384
  const int tid = threadIdx.x;
  for (int i = tid; i < Bsz * 48; i += 1024)
    ((uint4*)Hb)[i] = make_uint4(0, 0, 0, 0);
  stage_x(Xb0, xin, 0);                   // preload x_0
  stage_wait();
  v8f acc[3];
  int s = 0;
  for (int t = 0; t < Tsrc; ++t) {
    __bf16* Xcur = Xb0 + (size_t)s * (Bsz * Hd);        // arithmetic select
    __bf16* Xnxt = Xb0 + (size_t)(s ^ 1) * (Bsz * Hd);  // keeps AS(3) inference
    if (t + 1 < Tsrc) stage_x(Xnxt, xin, t + 1);        // overlap with WMMAs
    const __bf16* Wt = Wsw + launder_zero();
    v8f z = {};
    acc[0] = z; acc[1] = z; acc[2] = z;
    wg_gemm_part(Xcur, Hd, Wt, 0, 12, 24, acc);         // x @ Wih^T
    wg_gemm_part(Hb,   Hd, Wt, 12, 12, 24, acc);        // h @ Whh^T
    __syncthreads();                                    // old-h reads complete
    wg_epilogue(acc, bias, [&](int m, int n, float v) {
      const __bf16 hb = (__bf16)tanhf(v);
      Hb[m * Hd + n] = hb;                              // new h
      yout[((size_t)m * Tsrc + t) * Hd + n] = hb;       // layer output
    });
    stage_wait();                                       // tensorcnt + barrier
    s ^= 1;
  }
  for (int i = tid; i < Bsz * Hd; i += 1024) hfin[i] = Hb[i];
}

// ----------------- generic 64-row projection (no tanh) ---------------------
__global__ void __launch_bounds__(1024)
proj64_kernel(const __bf16* __restrict__ X, const __bf16* __restrict__ Wsw,
              const float* __restrict__ bias, float* __restrict__ out) {
  extern __shared__ char smem[];
  __bf16* A = (__bf16*)smem;              // 64 x 384
  const size_t row0 = (size_t)blockIdx.x * 64;
  const int tid = threadIdx.x;
  const __bf16* src = X + row0 * Hd;
  for (int i = tid; i < Bsz * 48; i += 1024)
    ((uint4*)A)[i] = ((const uint4*)src)[i];
  __syncthreads();
  v8f acc[3];
  v8f z = {};
  acc[0] = z; acc[1] = z; acc[2] = z;
  wg_gemm_part(A, Hd, Wsw, 0, 12, 12, acc);
  float* o = out + row0 * Hd;
  wg_epilogue(acc, bias, [&](int m, int n, float v) {
    o[(size_t)m * Hd + n] = v;
  });
}

// ------------------------- attention score pass ----------------------------
__global__ void __launch_bounds__(256)
dec_scores_kernel(const float* __restrict__ waq, const float* __restrict__ uk,
                  const float* __restrict__ VaW, const float* __restrict__ VaB,
                  float* __restrict__ scores) {
  const int gw = blockIdx.x * 8 + (threadIdx.x >> 5);   // 0..16383
  const int lane = threadIdx.x & 31;
  const int b = gw >> 8, t = gw & 255;
  const float* u = uk + ((size_t)b * Tsrc + t) * Hd;
  const float* q = waq + b * Hd;
  float s = 0.f;
  for (int k = lane; k < Hd; k += 32) s += VaW[k] * tanhf(q[k] + u[k]);
  for (int off = 16; off; off >>= 1) s += __shfl_xor(s, off, 32);
  if (!lane) scores[b * Tsrc + t] = s + VaB[0];
}

// ----------------- softmax + context + decoder input build -----------------
__global__ void __launch_bounds__(384)
dec_softctx_kernel(const float* __restrict__ scores,
                   const __bf16* __restrict__ keys,
                   const float* __restrict__ targets,
                   const float* __restrict__ eW, const float* __restrict__ eB,
                   __bf16* __restrict__ decA, int step) {
  __shared__ float sw[Tsrc];
  __shared__ float red[16];
  const int b = blockIdx.x, tid = threadIdx.x;
  const int lane = tid & 31, wv = tid >> 5;
  float v = (tid < Tsrc) ? scores[b * Tsrc + tid] : -3.4e38f;
  float m = v;
  for (int off = 16; off; off >>= 1) m = fmaxf(m, __shfl_xor(m, off, 32));
  if (!lane) red[wv] = m;
  __syncthreads();
  if (tid == 0) {
    float mm = red[0];
    for (int w = 1; w < 12; ++w) mm = fmaxf(mm, red[w]);
    red[0] = mm;
  }
  __syncthreads();
  const float mx = red[0];
  const float e = (tid < Tsrc) ? __expf(v - mx) : 0.f;
  if (tid < Tsrc) sw[tid] = e;
  float s = e;
  for (int off = 16; off; off >>= 1) s += __shfl_xor(s, off, 32);
  __syncthreads();
  if (!lane) red[wv] = s;
  __syncthreads();
  if (tid == 0) {
    float ss = 0.f;
    for (int w = 0; w < 12; ++w) ss += red[w];
    red[1] = ss;
  }
  __syncthreads();
  const float inv = 1.0f / red[1];
  float acc = 0.f;
  const __bf16* kp = keys + (size_t)b * Tsrc * Hd + tid;
  for (int t2 = 0; t2 < Tsrc; ++t2) acc += sw[t2] * (float)kp[(size_t)t2 * Hd];
  decA[b * 768 + 384 + tid] = (__bf16)(acc * inv);
  const float xv = (step == 0) ? -10001.0f : targets[(size_t)b * Tdec + step];
  decA[b * 768 + tid] = (__bf16)(xv * eW[tid] + eB[tid]);
}

// --------------------------- decoder RNN step ------------------------------
// One WG, 6 layers; TDM double-buffers next layer's hidden slab.
__global__ void __launch_bounds__(1024)
dec_rnn_kernel(const __bf16* __restrict__ decA, const __bf16* __restrict__ Wsw0,
               const __bf16* __restrict__ WswR, const float* __restrict__ bias6,
               __bf16* __restrict__ hidden, const float* __restrict__ headW,
               const float* __restrict__ headB, float* __restrict__ preds,
               int step) {
  extern __shared__ char smem[];
  __bf16* X   = (__bf16*)smem;            // 64 x 768 ([emb|ctx] then xl)
  __bf16* Hb0 = X + Bsz * 768;            // 64 x 384 (double buffered x2)
  const int tid = threadIdx.x;
  for (int i = tid; i < Bsz * 96; i += 1024)
    ((uint4*)X)[i] = ((const uint4*)decA)[i];
  stage_h(Hb0, hidden, 0);                // preload h_0
  stage_wait();
  v8f acc[3];
  int s = 0;
  for (int l = 0; l < Lnum; ++l) {
    __bf16* Hcur = Hb0 + (size_t)s * (Bsz * Hd);
    __bf16* Hnxt = Hb0 + (size_t)(s ^ 1) * (Bsz * Hd);
    if (l + 1 < Lnum) stage_h(Hnxt, hidden, l + 1);
    const __bf16* W =
        ((l == 0) ? Wsw0 : WswR + (size_t)(l - 1) * 768 * Hd) + launder_zero();
    const int KTfull = (l == 0) ? 36 : 24;
    const int KTx    = (l == 0) ? 24 : 12;
    v8f z = {};
    acc[0] = z; acc[1] = z; acc[2] = z;
    wg_gemm_part(X, 768, W, 0, KTx, KTfull, acc);       // xl @ Wih^T
    wg_gemm_part(Hcur, Hd, W, KTx, 12, KTfull, acc);    // h_l @ Whh^T
    __syncthreads();
    wg_epilogue(acc, bias6 + l * Hd, [&](int m, int n, float v) {
      const __bf16 hb = (__bf16)tanhf(v);
      X[m * 768 + n] = hb;                              // next layer input
      hidden[((size_t)l * Bsz + m) * Hd + n] = hb;
    });
    stage_wait();
    s ^= 1;
  }
  const int b = tid >> 4, li = tid & 15;
  float sacc = 0.f;
  for (int k = 0; k < 24; ++k) {
    const int h = li + k * 16;
    sacc += (float)X[b * 768 + h] * headW[h];
  }
  for (int off = 8; off; off >>= 1) sacc += __shfl_xor(sacc, off, 16);
  if (li == 0) preds[(size_t)b * Tdec + step] = sacc + headB[0];
}

// ------------------------------- launcher ----------------------------------
extern "C" void kernel_launch(void* const* d_in, const int* in_sizes, int n_in,
                              void* d_out, int out_size, void* d_ws,
                              size_t ws_size, hipStream_t stream) {
  (void)in_sizes; (void)n_in; (void)out_size; (void)ws_size;
  const float* inputs  = (const float*)d_in[0];
  const float* targets = (const float*)d_in[1];
  const float* embW    = (const float*)d_in[2];
  const float* embB    = (const float*)d_in[3];
  const float* WaW     = (const float*)d_in[4];
  const float* WaB     = (const float*)d_in[5];
  const float* UaW     = (const float*)d_in[6];
  const float* UaB     = (const float*)d_in[7];
  const float* VaW     = (const float*)d_in[8];
  const float* VaB     = (const float*)d_in[9];
  const float* encWih  = (const float*)d_in[10];
  const float* encWhh  = (const float*)d_in[11];
  const float* encBih  = (const float*)d_in[12];
  const float* encBhh  = (const float*)d_in[13];
  const float* decWih0 = (const float*)d_in[14];
  const float* decWih  = (const float*)d_in[15];
  const float* decWhh  = (const float*)d_in[16];
  const float* decBih  = (const float*)d_in[17];
  const float* decBhh  = (const float*)d_in[18];
  const float* headW   = (const float*)d_in[19];
  const float* headB   = (const float*)d_in[20];
  float* preds = (float*)d_out;

  char* ws = (char*)d_ws;
  auto alloc = [&](size_t bytes) {
    char* p = ws;
    ws += (bytes + 255) & ~(size_t)255;
    return p;
  };
  __bf16* encSw   = (__bf16*)alloc((size_t)Lnum * 768 * Hd * 2);
  __bf16* dec0Sw  = (__bf16*)alloc((size_t)1152 * Hd * 2);
  __bf16* decSw   = (__bf16*)alloc((size_t)(Lnum - 1) * 768 * Hd * 2);
  __bf16* uaSw    = (__bf16*)alloc((size_t)Hd * Hd * 2);
  __bf16* waSw    = (__bf16*)alloc((size_t)Hd * Hd * 2);
  float*  encBias = (float*)alloc((size_t)Lnum * Hd * 4);
  float*  decBias = (float*)alloc((size_t)Lnum * Hd * 4);
  __bf16* xbuf0   = (__bf16*)alloc((size_t)Bsz * Tsrc * Hd * 2);
  __bf16* xbuf1   = (__bf16*)alloc((size_t)Bsz * Tsrc * Hd * 2);
  float*  ukeys   = (float*)alloc((size_t)Bsz * Tsrc * Hd * 4);
  __bf16* hidBf   = (__bf16*)alloc((size_t)Lnum * Bsz * Hd * 2);
  float*  waq     = (float*)alloc((size_t)Bsz * Hd * 4);
  float*  scoresB = (float*)alloc((size_t)Bsz * Tsrc * 4);
  __bf16* decA    = (__bf16*)alloc((size_t)Bsz * 768 * 2);

  // ---- weight prep ----
  for (int l = 0; l < Lnum; ++l)
    swizzle_kernel<<<512, 256, 0, stream>>>(
        encWih + (size_t)l * Hd * Hd, encWhh + (size_t)l * Hd * Hd,
        Hd, Hd, Hd, encSw + (size_t)l * 768 * Hd);
  swizzle_kernel<<<512, 256, 0, stream>>>(decWih0, decWhh, 768, Hd, Hd, dec0Sw);
  for (int l = 1; l < Lnum; ++l)
    swizzle_kernel<<<512, 256, 0, stream>>>(
        decWih + (size_t)(l - 1) * Hd * Hd, decWhh + (size_t)l * Hd * Hd,
        Hd, Hd, Hd, decSw + (size_t)(l - 1) * 768 * Hd);
  swizzle_kernel<<<512, 256, 0, stream>>>(UaW, nullptr, Hd, 0, Hd, uaSw);
  swizzle_kernel<<<512, 256, 0, stream>>>(WaW, nullptr, Hd, 0, Hd, waSw);
  bias_add_kernel<<<9, 256, 0, stream>>>(encBih, encBhh, encBias, Lnum * Hd);
  bias_add_kernel<<<9, 256, 0, stream>>>(decBih, decBhh, decBias, Lnum * Hd);

  // ---- encoder ----
  embed_kernel<<<6144, 1024, 0, stream>>>(inputs, embW, embB, xbuf0);
  __bf16* xb[2] = {xbuf0, xbuf1};
  for (int l = 0; l < Lnum; ++l)
    enc_layer_kernel<<<1, 1024, 3 * Bsz * Hd * 2, stream>>>(
        xb[l & 1], xb[(l + 1) & 1], encSw + (size_t)l * 768 * Hd,
        encBias + l * Hd, hidBf + (size_t)l * Bsz * Hd);
  proj64_kernel<<<(Bsz * Tsrc) / 64, 1024, Bsz * Hd * 2, stream>>>(
      xbuf0, uaSw, UaB, ukeys);   // u_keys (step-invariant)

  // ---- decoder ----
  for (int i = 0; i < Tdec; ++i) {
    proj64_kernel<<<1, 1024, Bsz * Hd * 2, stream>>>(
        hidBf + (size_t)5 * Bsz * Hd, waSw, WaB, waq);
    dec_scores_kernel<<<(Bsz * Tsrc) / 8, 256, 0, stream>>>(
        waq, ukeys, VaW, VaB, scoresB);
    dec_softctx_kernel<<<Bsz, 384, 0, stream>>>(
        scoresB, xbuf0, targets, embW, embB, decA, i);
    dec_rnn_kernel<<<1, 1024, (Bsz * 768 + 2 * Bsz * Hd) * 2, stream>>>(
        decA, dec0Sw, decSw, decBias, hidBf, headW, headB, preds, i);
  }
}
